// CaserQueryEncoder_20547123544742
// MI455X (gfx1250) — matrix operand order repaired and checked by reading, not compiled, not run on hardware
//
#include <hip/hip_runtime.h>

// CaserQueryEncoder on MI455X (gfx1250, wave32):
//   ~93 GFLOP dominated by 50 variable-height horizontal convs -> bf16 WMMA.
//   v3: branch-free output-stationary t-banding. LDS weight slab is padded
//       with TB-1 zero rows on each side of the j-range so every band member
//       issues its WMMA unconditionally (OOB j multiplies zeros -> no-op).
//       This kills the exec-mask branches + accumulator phi-copies v2
//       suffered from, while keeping the ~3.5x A-fragment L2-traffic cut.
//       Small h (<12) takes a simple single-accumulator path.

#define B_   1024
#define L_   50
#define D_   128
#define NV_  8
#define NH_  16
#define OC_  (NV_*D_ + NH_*L_)   // 1824 = 57 * 32
#define TB_  4                   // t-band width (output-stationary tiles/wave)
#define ROWSP_MAX_ (L_ + 2*(TB_-1))            // 56 padded j-rows
#define SWPAD_BYTES_ (NH_ * ROWSP_MAX_ * D_ * 2)  // 229,376

typedef __attribute__((ext_vector_type(16))) __bf16 v16bf;
typedef __attribute__((ext_vector_type(8)))  float  v8f;

union Frag { v16bf v; uint4 q[2]; };

__device__ __forceinline__ unsigned short f2bf(float f) {
    unsigned int u = __float_as_uint(f);
    unsigned int r = u + 0x7FFFu + ((u >> 16) & 1u);   // round-nearest-even
    return (unsigned short)(r >> 16);
}
__device__ __forceinline__ float bf2f(unsigned short s) {
    return __uint_as_float(((unsigned int)s) << 16);
}

// ---------- conversion / gather kernels ----------

__global__ void k_cvt_w(const float* __restrict__ src,
                        unsigned short* __restrict__ dst, int n4) {
    int i = blockIdx.x * blockDim.x + threadIdx.x;
    if (i >= n4) return;
    float4 f = ((const float4*)src)[i];
    ushort4 o;
    o.x = f2bf(f.x); o.y = f2bf(f.y); o.z = f2bf(f.z); o.w = f2bf(f.w);
    ((ushort4*)dst)[i] = o;
}

// fc_w (K=1824, N=128) -> fcw_t (N=128, K=1824) in bf16 (K-contiguous B frags)
__global__ void k_cvt_fcw(const float* __restrict__ fc_w,
                          unsigned short* __restrict__ fcw_t) {
    int idx = blockIdx.x * blockDim.x + threadIdx.x;
    if (idx >= OC_ * D_) return;
    int k = idx / D_, n = idx % D_;
    fcw_t[n * OC_ + k] = f2bf(fc_w[idx]);
}

// E_bf[b,l,d] = bf16(item_emb[item_seq[b,l], d])
__global__ void k_gather_E(const int* __restrict__ item_seq,
                           const float* __restrict__ item_emb,
                           unsigned short* __restrict__ E_bf) {
    int idx = blockIdx.x * blockDim.x + threadIdx.x;
    if (idx >= B_ * L_ * D_) return;
    int row = idx / D_, d = idx % D_;
    int it = item_seq[row];
    E_bf[idx] = f2bf(item_emb[(size_t)it * D_ + d]);
}

// out[b, 128+d] = user_emb[user_ids[b], d]  (P_u half of the output, f32)
__global__ void k_gather_Pu(const int* __restrict__ user_ids,
                            const float* __restrict__ user_emb,
                            float* __restrict__ out) {
    int idx = blockIdx.x * blockDim.x + threadIdx.x;
    if (idx >= B_ * D_) return;
    int b = idx / D_, d = idx % D_;
    out[(size_t)b * 256 + 128 + d] = user_emb[(size_t)user_ids[b] * D_ + d];
}

// o_v[b, v*128+d] = sum_l E[b,l,d] * vfilter[l,v]   (0.1 GFLOP, VALU ok)
__global__ void k_vert(const unsigned short* __restrict__ E_bf,
                       const float* __restrict__ vfilter,
                       unsigned short* __restrict__ o_bf) {
    int idx = blockIdx.x * blockDim.x + threadIdx.x;
    if (idx >= B_ * NV_ * D_) return;
    int b = idx / (NV_ * D_);
    int r = idx % (NV_ * D_);
    int v = r / D_, d = r % D_;
    float acc = 0.f;
    const unsigned short* e = E_bf + (size_t)b * L_ * D_ + d;
#pragma unroll
    for (int l = 0; l < L_; ++l)
        acc += bf2f(e[l * D_]) * vfilter[l * NV_ + v];
    o_bf[(size_t)b * OC_ + v * D_ + d] = f2bf(acc);
}

#define WMMA_BF16(A, Bv, C) \
    __builtin_amdgcn_wmma_f32_16x16x32_bf16(false, (A), false, (Bv), (short)0, (C), false, false)

// ---------- horizontal convs: the 93-GFLOP WMMA core ----------
// grid = (50, 64), block = 256 (8 waves). blockIdx.x = i (h=i+1),
// blockIdx.y = b-tile (16 batches). LDS: W[i] compact [o][jp][d] with TB-1
// zero rows padded on each side of j (<=229.4KB) + 8KB max-reduction buffer.
// h>=12: waves take TB=4 t-bands, 4 accumulators, branch-free inner loop
// (each A fragment -> 4 WMMAs). h<12: simple single-accumulator path.
__global__ void __launch_bounds__(256)
k_hconv(const unsigned short* __restrict__ W_bf,   // (50,16,50,128) bf16
        const unsigned short* __restrict__ E_bf,   // (B,50,128) bf16
        const float* __restrict__ hconv_b,         // (50,16)
        unsigned short* __restrict__ o_bf)         // (B,1824) bf16
{
    extern __shared__ __align__(16) unsigned short sW[];
    float* sRed = (float*)((char*)sW + SWPAD_BYTES_);   // 8 waves x 256 f32
    const int i = blockIdx.x;
    const int h = i + 1;
    const int T = L_ - h + 1;
    const int rowsP = h + 2 * (TB_ - 1);     // padded j-extent

    // cooperative stage of zero-padded W[i], 16B per thread-chunk
    const int chunks = NH_ * rowsP * D_ / 8;
    for (int c = threadIdx.x; c < chunks; c += blockDim.x) {
        int e   = c * 8;
        int o   = e / (rowsP * D_);
        int rem = e - o * rowsP * D_;
        int jp  = rem / D_;
        int d   = rem - jp * D_;
        int j   = jp - (TB_ - 1);
        uint4 val = make_uint4(0u, 0u, 0u, 0u);
        if (j >= 0 && j < h)
            val = *(const uint4*)(W_bf + (((size_t)i * NH_ + o) * L_ + j) * D_ + d);
        ((uint4*)sW)[c] = val;
    }
    __syncthreads();

    const int lane = threadIdx.x & 31;
    const int wave = threadIdx.x >> 5;
    const int half = lane >> 4;       // K-half select per ISA A/B layouts
    const int mn   = lane & 15;       // A row m == B/C column n
    const int b0   = blockIdx.y * 16;

    const float bias = hconv_b[i * NH_ + mn];    // per output channel n

    const unsigned short* Abase = E_bf + (size_t)(b0 + mn) * L_ * D_;
    // points at j = 0 row of this lane's channel column
    const unsigned short* Bbase =
        sW + ((size_t)mn * rowsP + (TB_ - 1)) * D_ + half * 16;

    v8f runmax;
#pragma unroll
    for (int r = 0; r < 8; ++r) runmax[r] = 0.f;  // relu => max >= 0

    if (h >= 3 * TB_) {
        // -------- banded path: TB=4 output-stationary tiles per wave --------
        for (int t0 = wave * TB_; t0 < T; t0 += 8 * TB_) {
            int tmax = T - t0; if (tmax > TB_) tmax = TB_;
            v8f acc0 = {}, acc1 = {}, acc2 = {}, acc3 = {};
            const int send = t0 + tmax + h - 1;  // rows s = t0 .. send-1
            for (int s = t0; s < send; ++s) {
                const unsigned short* arow = Abase + (size_t)s * D_;
                __builtin_prefetch(arow + 2 * D_, 0, 0);
                const unsigned short* bj = Bbase + (size_t)(s - t0) * D_;
#pragma unroll
                for (int dc = 0; dc < 4; ++dc) { // K = 4 x 32 = 128 per row
                    Frag a;
                    a.q[0] = *(const uint4*)(arow + dc * 32 + half * 16);
                    a.q[1] = *(const uint4*)(arow + dc * 32 + half * 16 + 8);
                    Frag f0, f1, f2, f3;
                    const unsigned short* p = bj + dc * 32;
                    f0.q[0] = *(const uint4*)(p);
                    f0.q[1] = *(const uint4*)(p + 8);
                    f1.q[0] = *(const uint4*)(p - D_);
                    f1.q[1] = *(const uint4*)(p - D_ + 8);
                    f2.q[0] = *(const uint4*)(p - 2 * D_);
                    f2.q[1] = *(const uint4*)(p - 2 * D_ + 8);
                    f3.q[0] = *(const uint4*)(p - 3 * D_);
                    f3.q[1] = *(const uint4*)(p - 3 * D_ + 8);
                    acc0 = WMMA_BF16(a.v, f0.v, acc0);
                    acc1 = WMMA_BF16(a.v, f1.v, acc1);
                    acc2 = WMMA_BF16(a.v, f2.v, acc2);
                    acc3 = WMMA_BF16(a.v, f3.v, acc3);
                }
            }
            // fold band into running max (outside hot loop; tail guarded)
#pragma unroll
            for (int r = 0; r < 8; ++r) {
                float v0 = acc0[r] + bias; v0 = v0 > 0.f ? v0 : 0.f;
                runmax[r] = runmax[r] > v0 ? runmax[r] : v0;
            }
            if (tmax > 1) {
#pragma unroll
                for (int r = 0; r < 8; ++r) {
                    float v1 = acc1[r] + bias; v1 = v1 > 0.f ? v1 : 0.f;
                    runmax[r] = runmax[r] > v1 ? runmax[r] : v1;
                }
            }
            if (tmax > 2) {
#pragma unroll
                for (int r = 0; r < 8; ++r) {
                    float v2 = acc2[r] + bias; v2 = v2 > 0.f ? v2 : 0.f;
                    runmax[r] = runmax[r] > v2 ? runmax[r] : v2;
                }
            }
            if (tmax > 3) {
#pragma unroll
                for (int r = 0; r < 8; ++r) {
                    float v3 = acc3[r] + bias; v3 = v3 > 0.f ? v3 : 0.f;
                    runmax[r] = runmax[r] > v3 ? runmax[r] : v3;
                }
            }
        }
    } else {
        // -------- simple path: one accumulator, t strided across waves -----
        for (int t = wave; t < T; t += 8) {
            v8f acc = {};
            for (int j = 0; j < h; ++j) {
                const unsigned short* arow = Abase + (size_t)(t + j) * D_;
                const unsigned short* brow = Bbase + (size_t)j * D_;
#pragma unroll
                for (int dc = 0; dc < 4; ++dc) {
                    Frag a, b;
                    a.q[0] = *(const uint4*)(arow + dc * 32 + half * 16);
                    a.q[1] = *(const uint4*)(arow + dc * 32 + half * 16 + 8);
                    b.q[0] = *(const uint4*)(brow + dc * 32);
                    b.q[1] = *(const uint4*)(brow + dc * 32 + 8);
                    acc = WMMA_BF16(a.v, b.v, acc);
                }
            }
#pragma unroll
            for (int r = 0; r < 8; ++r) {
                float v = acc[r] + bias;
                v = v > 0.f ? v : 0.f;
                runmax[r] = runmax[r] > v ? runmax[r] : v;
            }
        }
    }

    // cross-wave max reduction through LDS
    // C/D layout: lane -> col n=mn, VGPR r -> row m = r + half*8
#pragma unroll
    for (int r = 0; r < 8; ++r)
        sRed[wave * 256 + (r + half * 8) * 16 + mn] = runmax[r];
    __syncthreads();
    {
        int m = threadIdx.x >> 4, n = threadIdx.x & 15;
        float v = sRed[threadIdx.x];
#pragma unroll
        for (int w = 1; w < 8; ++w)
            v = fmaxf(v, sRed[w * 256 + threadIdx.x]);
        o_bf[(size_t)(b0 + m) * OC_ + NV_ * D_ + i * NH_ + n] = f2bf(v);
    }
}

// ---------- FC: (1024 x 1824) @ (1824 x 128), bias + relu, WMMA ----------
// 512 tiles of 16x16; 8 waves/block, grid = 64. K = 57 exact x32 chunks.
__global__ void __launch_bounds__(256)
k_fc(const unsigned short* __restrict__ o_bf,    // (B,1824) bf16
     const unsigned short* __restrict__ fcw_t,   // (128,1824) bf16 (transposed)
     const float* __restrict__ fc_b,
     float* __restrict__ out)                    // (B,256) f32, cols 0..127
{
    const int lane = threadIdx.x & 31;
    const int wave = threadIdx.x >> 5;
    const int tile = blockIdx.x * 8 + wave;      // 0..511
    const int tm = tile / (D_ / 16);             // 0..63
    const int tn = tile % (D_ / 16);             // 0..7
    const int half = lane >> 4;
    const int mn   = lane & 15;
    const int b0 = tm * 16, n0 = tn * 16;

    const unsigned short* Abase = o_bf  + (size_t)(b0 + mn) * OC_;
    const unsigned short* Bbase = fcw_t + (size_t)(n0 + mn) * OC_ + half * 16;

    v8f acc = {};
    for (int kc = 0; kc < OC_ / 32; ++kc) {      // 57
        Frag a, b;
        const unsigned short* ap = Abase + kc * 32;
        const unsigned short* bp = Bbase + kc * 32;
        a.q[0] = *(const uint4*)(ap + half * 16);
        a.q[1] = *(const uint4*)(ap + half * 16 + 8);
        b.q[0] = *(const uint4*)(bp);
        b.q[1] = *(const uint4*)(bp + 8);
        acc = WMMA_BF16(a.v, b.v, acc);
    }
    const float bias = fc_b[n0 + mn];
#pragma unroll
    for (int r = 0; r < 8; ++r) {
        float v = acc[r] + bias;
        v = v > 0.f ? v : 0.f;
        int m = r + half * 8;
        out[(size_t)(b0 + m) * 256 + n0 + mn] = v;
    }
}

// ---------- launch ----------

extern "C" void kernel_launch(void* const* d_in, const int* in_sizes, int n_in,
                              void* d_out, int out_size, void* d_ws, size_t ws_size,
                              hipStream_t stream) {
    const int*   user_ids = (const int*)  d_in[0];
    const int*   item_seq = (const int*)  d_in[1];
    const float* user_emb = (const float*)d_in[2];
    const float* item_emb = (const float*)d_in[3];
    const float* vfilter  = (const float*)d_in[4];
    const float* hconv_w  = (const float*)d_in[5];
    const float* hconv_b  = (const float*)d_in[6];
    const float* fc_w     = (const float*)d_in[7];
    const float* fc_b     = (const float*)d_in[8];
    float* out = (float*)d_out;

    // workspace layout (bytes, 256-aligned)
    char* ws = (char*)d_ws;
    unsigned short* E_bf  = (unsigned short*)(ws + 0);          // 13,107,200
    unsigned short* W_bf  = (unsigned short*)(ws + 13107200);   // 10,240,000
    unsigned short* fcw_t = (unsigned short*)(ws + 23347200);   //    466,944
    unsigned short* o_bf  = (unsigned short*)(ws + 23814400);   //  3,735,552

    {   // hconv_w -> bf16 (5.12M elems, float4-vectorized)
        int n4 = L_ * NH_ * L_ * D_ / 4;
        k_cvt_w<<<(n4 + 255) / 256, 256, 0, stream>>>(hconv_w, W_bf, n4);
    }
    k_cvt_fcw<<<(OC_ * D_ + 255) / 256, 256, 0, stream>>>(fc_w, fcw_t);
    k_gather_E<<<(B_ * L_ * D_ + 255) / 256, 256, 0, stream>>>(item_seq, item_emb, E_bf);
    k_gather_Pu<<<(B_ * D_ + 255) / 256, 256, 0, stream>>>(user_ids, user_emb, out);
    k_vert<<<(B_ * NV_ * D_ + 255) / 256, 256, 0, stream>>>(E_bf, vfilter, o_bf);

    dim3 g3(L_, B_ / 16);                                // (i, b-tile)
    size_t lds = SWPAD_BYTES_ + 8 * 256 * sizeof(float); // 237,568 B
    k_hconv<<<g3, 256, lds, stream>>>(W_bf, E_bf, hconv_b, o_bf);

    k_fc<<<64, 256, 0, stream>>>(o_bf, fcw_t, fc_b, out);
}
